// Cross_Attention_1726576854484
// MI455X (gfx1250) — matrix operand order repaired and checked by reading, not compiled, not run on hardware
//
#include <hip/hip_runtime.h>
#include <math.h>

#define BATCH 8
#define NPOS  4096
#define DIM   512
#define HEADS 8
#define HK    64
#define EOUT  1024
#define EPSV  1e-5f

typedef __attribute__((ext_vector_type(16))) __bf16 v16bf;
typedef __attribute__((ext_vector_type(8)))  float  v8f;

union FragBF { unsigned u[8]; v16bf v; };

__device__ __forceinline__ unsigned short f2bf(float f) {
    union { float f; unsigned u; } c; c.f = f;
    unsigned r = c.u + 0x7FFFu + ((c.u >> 16) & 1u);   // round-to-nearest-even
    return (unsigned short)(r >> 16);
}
__device__ __forceinline__ unsigned pk2(float lo, float hi) {
    return (unsigned)f2bf(lo) | ((unsigned)f2bf(hi) << 16);
}

// gfx1250 async global->LDS copy (ASYNCcnt path). GV mode: %0 = LDS byte
// offset VGPR, %1 = 64-bit global address VGPR pair.
__device__ __forceinline__ void async_load_b128(unsigned lds_byte_off, const void* gaddr) {
    asm volatile("global_load_async_to_lds_b128 %0, %1, off"
                 :: "v"(lds_byte_off), "v"(gaddr) : "memory");
}
__device__ __forceinline__ void wait_async0() {
    asm volatile("s_wait_asynccnt 0x0" ::: "memory");
}
// Flat shared pointer -> wave-relative LDS byte offset (low 32 bits).
__device__ __forceinline__ unsigned lds_off(const void* p) {
    return (unsigned)(unsigned long long)p;
}

// ---------------------------------------------------------------------------
// K1a: per-(b,d) online softmax stats over N, split into 16 chunks of 256 n.
// ---------------------------------------------------------------------------
__global__ void key_stats_part(const float* __restrict__ x2,
                               float* __restrict__ pmax, float* __restrict__ psum) {
    int chunk = blockIdx.x;            // 0..15
    int b     = blockIdx.y;            // 0..7
    int d     = threadIdx.x;           // 0..511
    const float* p = x2 + ((size_t)b * NPOS + (size_t)chunk * 256) * DIM + d;
    float m = -3.4e38f, s = 0.f;
    for (int n = 0; n < 256; ++n) {
        float x  = p[(size_t)n * DIM];
        float nm = fmaxf(m, x);
        s = s * __expf(m - nm) + __expf(x - nm);
        m = nm;
    }
    size_t o = ((size_t)b * 16 + chunk) * DIM + d;
    pmax[o] = m;
    psum[o] = s;
}

// K1b: merge the 16 partial (max,sum) pairs per (b,d).
__global__ void key_stats_merge(const float* __restrict__ pmax, const float* __restrict__ psum,
                                float* __restrict__ colmax, float* __restrict__ colsum) {
    int b = blockIdx.x, d = threadIdx.x;
    float m = -3.4e38f;
    for (int c = 0; c < 16; ++c) m = fmaxf(m, pmax[((size_t)b * 16 + c) * DIM + d]);
    float s = 0.f;
    for (int c = 0; c < 16; ++c) {
        size_t o = ((size_t)b * 16 + c) * DIM + d;
        s += psum[o] * __expf(pmax[o] - m);
    }
    colmax[b * DIM + d] = m;
    colsum[b * DIM + d] = s;
}

// ---------------------------------------------------------------------------
// K2: query softmax over the 64 head channels, per (b,n,h); output bf16.
// ---------------------------------------------------------------------------
__global__ void query_softmax_kernel(const float* __restrict__ x2,
                                     unsigned short* __restrict__ qbf) {
    int t  = threadIdx.x;
    int r  = t >> 6;                    // row within block (0..3)
    int c  = t & 63;
    int h  = c >> 3;
    int s8 = c & 7;
    size_t bn   = (size_t)blockIdx.x * 4 + r;           // 0..32767
    size_t base = bn * DIM + h * HK + s8 * 8;
    const float4* p = (const float4*)(x2 + base);
    float4 a = p[0], q4 = p[1];
    float v[8] = {a.x, a.y, a.z, a.w, q4.x, q4.y, q4.z, q4.w};
    float m = v[0];
#pragma unroll
    for (int i = 1; i < 8; ++i) m = fmaxf(m, v[i]);
    m = fmaxf(m, __shfl_xor(m, 1, 8));
    m = fmaxf(m, __shfl_xor(m, 2, 8));
    m = fmaxf(m, __shfl_xor(m, 4, 8));
    float s = 0.f;
#pragma unroll
    for (int i = 0; i < 8; ++i) { v[i] = __expf(v[i] - m); s += v[i]; }
    s += __shfl_xor(s, 1, 8);
    s += __shfl_xor(s, 2, 8);
    s += __shfl_xor(s, 4, 8);
    float inv = 1.f / s;
    unsigned* q32 = (unsigned*)(qbf + base);
#pragma unroll
    for (int i = 0; i < 4; ++i) q32[i] = pk2(v[2 * i] * inv, v[2 * i + 1] * inv);
}

// ---------------------------------------------------------------------------
// K3: context[b,h,k,v] = sum_n key[k,n]*val[v,n]  via WMMA bf16, K=4096.
// ---------------------------------------------------------------------------
__global__ void context_kernel(const float* __restrict__ x1, const float* __restrict__ x2,
                               const float* __restrict__ colmax, const float* __restrict__ colsum,
                               float* __restrict__ ctx) {
    __shared__ unsigned short keyT[64 * 32];   // [channel k][n]
    __shared__ unsigned short valT[64 * 32];   // [channel v][n]
    int bh = blockIdx.x;
    int b = bh >> 3, h = bh & 7;
    int t = threadIdx.x;
    int c = t & 63, j4 = t >> 6;               // loader mapping
    int lane = t & 31, wave = t >> 5;
    int kt  = wave >> 1;                        // k-tile (0..3)
    int vt0 = (wave & 1) * 2;                   // first v-tile
    int m = lane & 15, hi = lane >> 4;

    float cm = colmax[b * DIM + h * HK + c];
    float rz = 1.f / colsum[b * DIM + h * HK + c];
    const float* p2 = x2 + (size_t)b * NPOS * DIM + h * HK + c;
    const float* p1 = x1 + (size_t)b * NPOS * DIM + h * HK + c;

    v8f acc0, acc1;
#pragma unroll
    for (int i = 0; i < 8; ++i) { acc0[i] = 0.f; acc1[i] = 0.f; }

    for (int n0 = 0; n0 < NPOS; n0 += 32) {
        __syncthreads();
#pragma unroll
        for (int jj = 0; jj < 8; ++jj) {
            int j = jj * 4 + j4;
            size_t off = (size_t)(n0 + j) * DIM;
            keyT[c * 32 + j] = f2bf(__expf(p2[off] - cm) * rz);
            valT[c * 32 + j] = f2bf(p1[off]);
        }
        __syncthreads();
        FragBF fa, fb0, fb1;
#pragma unroll
        for (int i = 0; i < 8; ++i) {
            int kb = hi * 16 + 2 * i;
            fa.u[i]  = *(const unsigned*)&keyT[(kt * 16 + m) * 32 + kb];
            fb0.u[i] = *(const unsigned*)&valT[(vt0 * 16 + m) * 32 + hi * 16 + 2 * i];
            fb1.u[i] = *(const unsigned*)&valT[((vt0 + 1) * 16 + m) * 32 + hi * 16 + 2 * i];
        }
        acc0 = __builtin_amdgcn_wmma_f32_16x16x32_bf16(false, fa.v, false, fb0.v,
                                                       (short)0, acc0, false, false);
        acc1 = __builtin_amdgcn_wmma_f32_16x16x32_bf16(false, fa.v, false, fb1.v,
                                                       (short)0, acc1, false, false);
    }
    float* cbase = ctx + (size_t)bh * 64 * 64;
#pragma unroll
    for (int i = 0; i < 8; ++i) {
        int k = kt * 16 + i + hi * 8;
        cbase[k * 64 + vt0 * 16 + m]       = acc0[i];
        cbase[k * 64 + (vt0 + 1) * 16 + m] = acc1[i];
    }
}

// ---------------------------------------------------------------------------
// K4: fold context into reprojection weight:
//     Wc[b,e,h*64+k] = sum_v W_rep[e,h*64+v]*ctx[b,h,k,v]   (bf16 output)
//     Context tile staged via async global->LDS (16KB straight copy).
// ---------------------------------------------------------------------------
__global__ void wc_kernel(const float* __restrict__ Wrep, const float* __restrict__ ctx,
                          unsigned short* __restrict__ wc) {
    __shared__ float ctxS[64 * 64];
    __shared__ float Ws[4 * 64];
    int bh = blockIdx.x;
    int b = bh >> 3, h = bh & 7;
    int t = threadIdx.x;
    const float* cbase = ctx + (size_t)bh * 4096;
    {
        unsigned ldsbase = lds_off(ctxS);
        const char* gsrc = (const char*)cbase;
#pragma unroll
        for (int i = 0; i < 4; ++i)
            async_load_b128(ldsbase + t * 16 + i * 4096, gsrc + t * 16 + i * 4096);
        wait_async0();   // own async ops done; loop-top barrier publishes to block
    }
    int k = t & 63, eg = t >> 6;
    for (int e0 = 0; e0 < EOUT; e0 += 4) {
        __syncthreads();                 // WAR on Ws + covers async ctxS fill
        Ws[t] = Wrep[(size_t)(e0 + eg) * DIM + h * HK + (t & 63)];
        __syncthreads();
        float acc = 0.f;
#pragma unroll 8
        for (int v = 0; v < 64; ++v) acc += Ws[eg * 64 + v] * ctxS[k * 64 + v];
        wc[((size_t)b * EOUT + e0 + eg) * DIM + h * HK + k] = f2bf(acc);
    }
}

// ---------------------------------------------------------------------------
// K5: Y[b,n,e] = sum_d q[b,n,d]*Wc[b,e,d] + bias, then LayerNorm over e.
//     A-tile staged via async global->LDS; Wc stream prefetched.
// ---------------------------------------------------------------------------
__global__ void gemm_ln_kernel(const unsigned short* __restrict__ qbf,
                               const unsigned short* __restrict__ wc,
                               const float* __restrict__ brep,
                               const float* __restrict__ lnw,
                               const float* __restrict__ lnb,
                               float* __restrict__ out) {
    __shared__ union { unsigned short a[16 * DIM]; float y[16 * EOUT]; } sm;
    int b  = blockIdx.x >> 8;
    int n0 = (blockIdx.x & 255) * 16;
    int t = threadIdx.x, lane = t & 31, wave = t >> 5;
    int m = lane & 15, hi = lane >> 4;

    {   // stage 16x512 bf16 A tile into LDS via async copy (4KB per step)
        unsigned ldsbase = lds_off(sm.a);
        const char* gsrc = (const char*)(qbf + ((size_t)b * NPOS + n0) * DIM);
#pragma unroll
        for (int i = 0; i < 4; ++i)
            async_load_b128(ldsbase + t * 16 + i * 4096, gsrc + t * 16 + i * 4096);
        wait_async0();
    }
    __syncthreads();

    int ebase = wave * 128;
    v8f acc[8];
#pragma unroll
    for (int et = 0; et < 8; ++et)
#pragma unroll
        for (int i = 0; i < 8; ++i) acc[et][i] = 0.f;

    const unsigned* wc32 = (const unsigned*)wc;
    size_t wrowBase = ((size_t)b * EOUT + ebase + m) * DIM;

    for (int kc = 0; kc < DIM; kc += 32) {
        if (kc + 32 < DIM)   // pull next k-chunk of the Wc stream toward L2/L0
            __builtin_prefetch((const void*)&wc32[(wrowBase + kc + 32 + hi * 16) >> 1], 0, 0);
        FragBF fa;
#pragma unroll
        for (int i = 0; i < 8; ++i) {
            int kb = hi * 16 + 2 * i;
            fa.u[i] = *(const unsigned*)&sm.a[m * DIM + kc + kb];
        }
#pragma unroll
        for (int et = 0; et < 8; ++et) {
            FragBF fb;
            size_t base16 = wrowBase + (size_t)et * 16 * DIM + kc + hi * 16;
#pragma unroll
            for (int i = 0; i < 8; ++i) fb.u[i] = wc32[(base16 >> 1) + i];
            acc[et] = __builtin_amdgcn_wmma_f32_16x16x32_bf16(false, fa.v, false, fb.v,
                                                              (short)0, acc[et], false, false);
        }
    }
    __syncthreads();   // all waves done reading sm.a before overwriting with y

#pragma unroll
    for (int et = 0; et < 8; ++et) {
        int e = ebase + et * 16 + m;
        float bias = brep[e];
#pragma unroll
        for (int i = 0; i < 8; ++i) {
            int row = i + hi * 8;
            sm.y[row * EOUT + e] = acc[et][i] + bias;
        }
    }
    __syncthreads();

    // LayerNorm: each wave owns 2 rows; lane-parallel over 1024 e.
    for (int rr = 0; rr < 2; ++rr) {
        int row = wave * 2 + rr;
        float s = 0.f, sq = 0.f;
#pragma unroll
        for (int j = 0; j < 32; ++j) {
            float v = sm.y[row * EOUT + lane + 32 * j];
            s += v; sq += v * v;
        }
#pragma unroll
        for (int msk = 16; msk >= 1; msk >>= 1) {
            s  += __shfl_xor(s,  msk, 32);
            sq += __shfl_xor(sq, msk, 32);
        }
        float mu  = s * (1.f / EOUT);
        float var = sq * (1.f / EOUT) - mu * mu;
        float rs  = rsqrtf(var + EPSV);
        float* orow = out + ((size_t)b * NPOS + n0 + row) * EOUT;
#pragma unroll
        for (int j = 0; j < 32; ++j) {
            int e = lane + 32 * j;
            float v = sm.y[row * EOUT + e];
            orow[e] = (v - mu) * rs * lnw[e] + lnb[e];
        }
    }
}

// ---------------------------------------------------------------------------
extern "C" void kernel_launch(void* const* d_in, const int* in_sizes, int n_in,
                              void* d_out, int out_size, void* d_ws, size_t ws_size,
                              hipStream_t stream) {
    (void)in_sizes; (void)n_in; (void)out_size; (void)ws_size;
    const float* x1   = (const float*)d_in[0];
    const float* x2   = (const float*)d_in[1];
    const float* Wrep = (const float*)d_in[2];
    const float* brep = (const float*)d_in[3];
    const float* lnw  = (const float*)d_in[4];
    const float* lnb  = (const float*)d_in[5];
    float* out = (float*)d_out;

    // Workspace layout (~43.5 MB)
    float* ctx    = (float*)d_ws;                                 // 8*8*64*64 f32
    float* colmax = ctx + (size_t)BATCH * HEADS * HK * HK;        // 4096 f32
    float* colsum = colmax + BATCH * DIM;                         // 4096 f32
    float* pmax   = colsum + BATCH * DIM;                         // 8*16*512 f32
    float* psum   = pmax + (size_t)BATCH * 16 * DIM;              // 8*16*512 f32
    unsigned short* wc  = (unsigned short*)(psum + (size_t)BATCH * 16 * DIM); // 8*1024*512 bf16
    unsigned short* qbf = wc + (size_t)BATCH * EOUT * DIM;        // 8*4096*512 bf16

    key_stats_part <<<dim3(16, BATCH), DIM, 0, stream>>>(x2, pmax, psum);
    key_stats_merge<<<BATCH, DIM, 0, stream>>>(pmax, psum, colmax, colsum);
    query_softmax_kernel<<<(BATCH * NPOS) / 4, 256, 0, stream>>>(x2, qbf);
    context_kernel<<<BATCH * HEADS, 256, 0, stream>>>(x1, x2, colmax, colsum, ctx);
    wc_kernel<<<BATCH * HEADS, 256, 0, stream>>>(Wrep, ctx, wc);
    gemm_ln_kernel<<<BATCH * (NPOS / 16), 256, 0, stream>>>(qbf, wc, brep, lnw, lnb, out);
}
